// AttentionLayer_64974265254274
// MI455X (gfx1250) — matrix operand order repaired and checked by reading, not compiled, not run on hardware
//
#include <hip/hip_runtime.h>

// ---------------------------------------------------------------------------
// MI455X (gfx1250) causal attention: W pre-convert + QKV projection + flash
// attention.  bf16 WMMA (v_wmma_f32_16x16x32_bf16), fp32 accumulate,
// fp32 softmax statistics.
// ---------------------------------------------------------------------------

#define DMODEL 1024
#define DK     64
#define SEQ    4096
#define BATCH  4
#define NROWS  (BATCH * SEQ)   // 16384

typedef __attribute__((ext_vector_type(16))) __bf16 v16bf;
typedef __attribute__((ext_vector_type(8)))  float  v8f;
typedef unsigned int  v4u  __attribute__((ext_vector_type(4), aligned(16), __may_alias__));
typedef float         v4f  __attribute__((ext_vector_type(4), aligned(16), __may_alias__));
typedef __bf16        v4bf __attribute__((ext_vector_type(4), aligned(8),  __may_alias__));

union FragBF {
    v16bf v;
    unsigned short u[16];
    v4u q[2];
};

__device__ __forceinline__ unsigned short f32_to_bf16(float f) {
    unsigned int b = __builtin_bit_cast(unsigned int, f);
    b += 0x7FFFu + ((b >> 16) & 1u);          // round-to-nearest-even
    return (unsigned short)(b >> 16);
}

__device__ __forceinline__ v8f zero8() {
    v8f z;
#pragma unroll
    for (int i = 0; i < 8; ++i) z[i] = 0.0f;
    return z;
}

// ---------------------------------------------------------------------------
// Kernel 0: one-time weight conversion fp32 -> bf16, transposed to col-major
// Wt[w][col][k] so projection B-fragments are contiguous 32B loads.
// ---------------------------------------------------------------------------
__global__ __launch_bounds__(256) void convert_w_kernel(
    const float* __restrict__ Wq,
    const float* __restrict__ Wk,
    const float* __restrict__ Wv,
    unsigned short* __restrict__ Wt)   // [3][64][1024] bf16
{
    const int idx = blockIdx.x * 256 + threadIdx.x;   // 0 .. 3*64*1024-1
    const int w   = idx >> 16;
    const int c   = (idx >> 10) & 63;
    const int k   = idx & 1023;
    const float* W = (w == 0) ? Wq : (w == 1) ? Wk : Wv;
    Wt[idx] = f32_to_bf16(W[(size_t)k * DK + c]);
}

// ---------------------------------------------------------------------------
// Kernel 1: Q,K,V projections.  Grid: 1024 blocks (one 16-row M tile each),
// 128 threads = 4 waves; wave w computes 16-column N tile w for Q, K, V.
// The x A-tile is converted to bf16 ONCE per block into LDS (cooperative),
// then each wave reads its A fragment with two ds_load_b128.
// Q is pre-scaled by 1/sqrt(DK).  V is stored transposed (d-major).
// ---------------------------------------------------------------------------
__global__ __launch_bounds__(128) void qkv_proj_kernel(
    const float* __restrict__ x,
    const unsigned short* __restrict__ Wt,   // [3][64][1024] bf16
    unsigned short* __restrict__ Qb,         // [NROWS][64] bf16 (pre-scaled)
    unsigned short* __restrict__ Kb,         // [NROWS][64] bf16
    unsigned short* __restrict__ Vt)         // [BATCH][64][SEQ] bf16
{
    __shared__ unsigned short xlds[16 * 40];     // 16 rows x 32 k, padded stride

    const int lane = threadIdx.x & 31;
    const int wave = threadIdx.x >> 5;
    const int hi   = lane >> 4;
    const int l15  = lane & 15;
    const int m0   = blockIdx.x * 16;
    const int n0   = wave * 16;

    // cooperative staging coordinates: 512 elements / 128 threads = 4 each
    const int srow = threadIdx.x >> 3;           // 0..15
    const int skl  = (threadIdx.x & 7) * 4;      // 0,4,..,28

    // per-wave weight column bases (col-major bf16)
    const int c = n0 + l15;
    const unsigned short* Wtq = Wt + ((size_t)0 * DK + c) * DMODEL;
    const unsigned short* Wtk = Wt + ((size_t)1 * DK + c) * DMODEL;
    const unsigned short* Wtv = Wt + ((size_t)2 * DK + c) * DMODEL;

    v8f accq = zero8(), acck = zero8(), accv = zero8();

    for (int k0 = 0; k0 < DMODEL; k0 += 32) {
        // --- stage 16x32 x-tile into LDS as bf16 (converted once) -----------
        {
            v4f f = *(const v4f*)(x + (size_t)(m0 + srow) * DMODEL + k0 + skl);
            v4bf b = __builtin_convertvector(f, v4bf);
            *(v4bf*)(xlds + srow * 40 + skl) = b;
        }
        __syncthreads();

        // --- A fragment from LDS: lane row = l15 ----------------------------
        FragBF a;
        {
            const v4u* p = (const v4u*)(xlds + l15 * 40);
            a.q[0] = p[2 * hi];        // k = hi*16 .. hi*16+7
            a.q[1] = p[2 * hi + 1];    // k = hi*16+8 .. hi*16+15
        }

        // --- B fragments: contiguous 32B loads from col-major Wt ------------
        FragBF bq, bk, bv;
        {
            const v4u* pq = (const v4u*)(Wtq + k0 + hi * 16);
            const v4u* pk = (const v4u*)(Wtk + k0 + hi * 16);
            const v4u* pv = (const v4u*)(Wtv + k0 + hi * 16);
            bq.q[0] = pq[0]; bq.q[1] = pq[1];
            bk.q[0] = pk[0]; bk.q[1] = pk[1];
            bv.q[0] = pv[0]; bv.q[1] = pv[1];
        }

        accq = __builtin_amdgcn_wmma_f32_16x16x32_bf16(false, a.v, false, bq.v,
                                                       (short)0, accq, false, false);
        acck = __builtin_amdgcn_wmma_f32_16x16x32_bf16(false, a.v, false, bk.v,
                                                       (short)0, acck, false, false);
        accv = __builtin_amdgcn_wmma_f32_16x16x32_bf16(false, a.v, false, bv.v,
                                                       (short)0, accv, false, false);
        __syncthreads();
    }

    // --- store: C/D layout lane -> (row = m0 + hi*8 + r, col = n0 + l15) ----
    const float qscale = 0.125f;  // 1/sqrt(64)
#pragma unroll
    for (int r = 0; r < 8; ++r) {
        const int row = m0 + hi * 8 + r;
        const int col = n0 + l15;
        Qb[(size_t)row * DK + col] = f32_to_bf16(accq[r] * qscale);
        Kb[(size_t)row * DK + col] = f32_to_bf16(acck[r]);
        const int b = row >> 12;          // /SEQ
        const int s = row & (SEQ - 1);
        Vt[((size_t)b * DK + col) * SEQ + s] = f32_to_bf16(accv[r]);
    }
}

// ---------------------------------------------------------------------------
// Kernel 2: causal flash attention, one 16-query tile per wave.
// Grid: 256 blocks x 128 threads (4 waves) = 1024 query tiles.
// Key blocks of 32: QK^T = 4 WMMA, PV = 4 WMMA; online softmax in fp32.
// ---------------------------------------------------------------------------
__global__ __launch_bounds__(128) void flash_attn_kernel(
    const unsigned short* __restrict__ Qb,
    const unsigned short* __restrict__ Kb,
    const unsigned short* __restrict__ Vt,
    float* __restrict__ out)           // [BATCH][SEQ][64] fp32
{
    __shared__ unsigned short plds[4 * 16 * 40];   // wave-private P tiles (padded)

    const int lane = threadIdx.x & 31;
    const int wave = threadIdx.x >> 5;
    const int hi   = lane >> 4;
    const int l15  = lane & 15;

    const int tile = blockIdx.x * 4 + wave;        // 0..1023
    const int b    = tile >> 8;                    // /(SEQ/16)
    const int q0   = (tile & 255) << 4;            // query tile start within batch

    const unsigned short* Qbase = Qb + (size_t)b * SEQ * DK;
    const unsigned short* Kbase = Kb + (size_t)b * SEQ * DK;
    const unsigned short* Vtb   = Vt + (size_t)b * DK * SEQ;
    unsigned short* lp = plds + wave * 16 * 40;

    // --- Q fragments (resident): A layout over d, two k-steps ---------------
    FragBF qf[2];
#pragma unroll
    for (int kk = 0; kk < 2; ++kk) {
        const v4u* p0 = (const v4u*)(Qbase + (size_t)(q0 + l15) * DK + kk * 32 + hi * 16);
        const v4u* p1 = (const v4u*)(Qbase + (size_t)(q0 + l15) * DK + kk * 32 + hi * 16 + 8);
        qf[kk].q[0] = p0[0];
        qf[kk].q[1] = p1[0];
    }

    float mrow[8], lrow[8];
    v8f   o[4];
#pragma unroll
    for (int r = 0; r < 8; ++r) { mrow[r] = -1e30f; lrow[r] = 0.0f; }
#pragma unroll
    for (int f = 0; f < 4; ++f) o[f] = zero8();

    for (int s0 = 0; s0 < q0 + 16; s0 += 32) {
        // prefetch next key block into cache (global_prefetch_b8)
        __builtin_prefetch(Kbase + (size_t)(s0 + 32 + l15) * DK, 0, 3);
        __builtin_prefetch(Vtb + (size_t)l15 * SEQ + s0 + 32, 0, 3);

        // --- scores S = (Q*scale) K^T : two 16x16 tiles ---------------------
        v8f sc[2];
        sc[0] = zero8(); sc[1] = zero8();
#pragma unroll
        for (int t = 0; t < 2; ++t) {
            const int key = s0 + t * 16 + l15;     // B-frag column
#pragma unroll
            for (int kk = 0; kk < 2; ++kk) {
                FragBF kfrag;   // element i -> d = kk*32 + hi*16 + i (contiguous)
                const v4u* p = (const v4u*)(Kbase + (size_t)key * DK + kk * 32 + hi * 16);
                kfrag.q[0] = p[0];
                kfrag.q[1] = p[1];
                sc[t] = __builtin_amdgcn_wmma_f32_16x16x32_bf16(
                    false, qf[kk].v, false, kfrag.v, (short)0, sc[t], false, false);
            }
        }

        // --- causal mask (only the final block can cross the diagonal) ------
        const bool needMask = (s0 + 31 > q0);
        if (needMask) {
#pragma unroll
            for (int r = 0; r < 8; ++r) {
                const int qrow = q0 + hi * 8 + r;
#pragma unroll
                for (int t = 0; t < 2; ++t) {
                    const int key = s0 + t * 16 + l15;
                    if (key > qrow) sc[t][r] = -1e30f;
                }
            }
        }

        // --- online softmax: row max / exp / row sum / rescale O ------------
#pragma unroll
        for (int r = 0; r < 8; ++r) {
            float v = fmaxf(sc[0][r], sc[1][r]);
#pragma unroll
            for (int m = 1; m < 16; m <<= 1) v = fmaxf(v, __shfl_xor(v, m, 32));
            const float newm = fmaxf(mrow[r], v);

            float rs = 0.0f;
#pragma unroll
            for (int t = 0; t < 2; ++t) {
                const float p = __expf(sc[t][r] - newm);
                rs += p;
                lp[(hi * 8 + r) * 40 + t * 16 + l15] = f32_to_bf16(p);
            }
#pragma unroll
            for (int m = 1; m < 16; m <<= 1) rs += __shfl_xor(rs, m, 32);

            const float corr = __expf(mrow[r] - newm);
            lrow[r] = lrow[r] * corr + rs;
            mrow[r] = newm;
#pragma unroll
            for (int f = 0; f < 4; ++f) o[f][r] *= corr;
        }

        // --- reload P as an A fragment (LDS transpose, wave-private) --------
        FragBF pf;
        {
            const v4u* p = (const v4u*)(lp + l15 * 40);  // row = l15, 80B stride
            pf.q[0] = p[2 * hi];        // elements 0..7  -> keys hi*16 .. hi*16+7
            pf.q[1] = p[2 * hi + 1];    // elements 8..15 -> keys hi*16+8 .. hi*16+15
        }

        // --- O += P * V  (V stored d-major: contiguous B fragments) ---------
#pragma unroll
        for (int f = 0; f < 4; ++f) {
            FragBF vfrag;  // column d = f*16 + l15; element i -> key s0 + hi*16 + i
            const v4u* p = (const v4u*)(Vtb + (size_t)(f * 16 + l15) * SEQ + s0 + hi * 16);
            vfrag.q[0] = p[0];
            vfrag.q[1] = p[1];
            o[f] = __builtin_amdgcn_wmma_f32_16x16x32_bf16(
                false, pf.v, false, vfrag.v, (short)0, o[f], false, false);
        }
    }

    // --- epilogue: y = O / l ------------------------------------------------
#pragma unroll
    for (int r = 0; r < 8; ++r) {
        const int row   = q0 + hi * 8 + r;
        const float inv = 1.0f / lrow[r];
#pragma unroll
        for (int f = 0; f < 4; ++f) {
            out[((size_t)b * SEQ + row) * DK + f * 16 + l15] = o[f][r] * inv;
        }
    }
}

// ---------------------------------------------------------------------------
extern "C" void kernel_launch(void* const* d_in, const int* in_sizes, int n_in,
                              void* d_out, int out_size, void* d_ws, size_t ws_size,
                              hipStream_t stream)
{
    (void)in_sizes; (void)n_in; (void)out_size; (void)ws_size;
    const float* x  = (const float*)d_in[0];
    const float* Wq = (const float*)d_in[1];
    const float* Wk = (const float*)d_in[2];
    const float* Wv = (const float*)d_in[3];

    unsigned short* Qb = (unsigned short*)d_ws;            // 16384*64 bf16
    unsigned short* Kb = Qb + (size_t)NROWS * DK;          // 16384*64 bf16
    unsigned short* Vt = Kb + (size_t)NROWS * DK;          // 4*64*4096 bf16
    unsigned short* Wt = Vt + (size_t)BATCH * DK * SEQ;    // 3*64*1024 bf16

    convert_w_kernel<<<(3 * DK * DMODEL) / 256, 256, 0, stream>>>(Wq, Wk, Wv, Wt);
    qkv_proj_kernel<<<NROWS / 16, 128, 0, stream>>>(x, Wt, Qb, Kb, Vt);
    flash_attn_kernel<<<(NROWS / 16) / 4, 128, 0, stream>>>(Qb, Kb, Vt, (float*)d_out);
}
